// SE_Fusion_Four_19292993093896
// MI455X (gfx1250) — matrix operand order repaired and checked by reading, not compiled, not run on hardware
//
#include <hip/hip_runtime.h>

// ---------------------------------------------------------------------------
// SE_Fusion_Four on MI455X (gfx1250, wave32).
//
// A = 0.25*ones(4,4) exactly (K4 + self loops, deg=4 everywhere), so:
//     t      = (0.25*(fea1+fea2+fea3+fea4)) @ Wg + bg            [B,512]
//     hid[k] = relu(t @ W1s[k] + b1[k]),  W1s[k] = sum of W1[k] row blocks
//     out_k  = relu(sigmoid(hid[k] @ W2[k] + b2[k])) * fea_k
// All GEMMs via v_wmma_f32_16x16x32_bf16; FC1+FC2 fused per head with the
// hid tile resident in LDS (saves ~134 MB of HBM traffic ~ 5.7us @ 23.3TB/s).
// Tiles staged with GLOBAL_LOAD_ASYNC_TO_LDS_B128 (ASYNCcnt/s_wait_asynccnt).
// ---------------------------------------------------------------------------

typedef __attribute__((ext_vector_type(16))) __bf16 v16bf;
typedef __attribute__((ext_vector_type(8)))  float  v8f;
typedef int v4i_ __attribute__((vector_size(16)));   // matches builtin proto

union FragBF { uint4 u[2]; v16bf v; };

__device__ __forceinline__ unsigned short f2bf(float f) {
  unsigned u = __builtin_bit_cast(unsigned, f);
  u += 0x7FFFu + ((u >> 16) & 1u);          // round to nearest even
  return (unsigned short)(u >> 16);
}

static constexpr int Bsz = 16384;
static constexpr int C   = 512;

// --------------------------- async LDS staging ------------------------------
#if defined(__has_builtin)
#if __has_builtin(__builtin_amdgcn_global_load_async_to_lds_b128) && \
    __has_builtin(__builtin_amdgcn_s_wait_asynccnt)
#define USE_ASYNC_LDS 1
#endif
#endif

#ifdef USE_ASYNC_LDS
// arg0: global (AS1) int4*, arg1: LDS (AS3) int4*, imm offset, imm cpol
#define CP16(ldst, gsrc)                                                  \
  __builtin_amdgcn_global_load_async_to_lds_b128(                         \
      (__attribute__((address_space(1))) v4i_*)(gsrc),                    \
      (__attribute__((address_space(3))) v4i_*)(ldst), 0, 0)
#define WAIT_STAGE() __builtin_amdgcn_s_wait_asynccnt(0)
#else
#define CP16(ldst, gsrc) (*(uint4*)(ldst) = *(const uint4*)(gsrc))
#define WAIT_STAGE()
#endif

// ---------------------------------------------------------------------------
// Weight prep: fold W1 node-blocks, transpose everything to [N][K] bf16 so a
// WMMA B-fragment (32x16, K-major per lane) is two 16-byte LDS reads per lane.
// idx space: 9 matrices * 512*512.  mat 0 = Wg, 1..4 = W1s[k], 5..8 = W2[k].
// ---------------------------------------------------------------------------
__global__ __launch_bounds__(256) void prep_weights_kernel(
    const float* __restrict__ Wg, const float* __restrict__ W1,
    const float* __restrict__ W2,
    unsigned short* __restrict__ Wgb, unsigned short* __restrict__ W1sb,
    unsigned short* __restrict__ W2b) {
  int idx = blockIdx.x * 256 + threadIdx.x;
  int mat = idx >> 18;              // 512*512 = 2^18
  int rem = idx & 262143;
  int n = rem >> 9;                 // output (N) index
  int k = rem & 511;                // reduction (K) index
  if (mat == 0) {
    Wgb[n * 512 + k] = f2bf(Wg[k * 512 + n]);
  } else if (mat < 5) {
    int h = mat - 1;
    const float* base = W1 + (size_t)h * 2048 * 512;
    float s = 0.f;
#pragma unroll
    for (int blk = 0; blk < 4; ++blk)
      s += base[(size_t)(blk * 512 + k) * 512 + n];
    W1sb[(size_t)h * 262144 + n * 512 + k] = f2bf(s);
  } else {
    int h = mat - 5;
    W2b[(size_t)h * 262144 + n * 512 + k] =
        f2bf(W2[((size_t)h * 512 + k) * 512 + n]);
  }
}

// ---------------------------------------------------------------------------
// s = bf16(0.25*(f1+f2+f3+f4))
// ---------------------------------------------------------------------------
__global__ __launch_bounds__(256) void mix_kernel(
    const float4* __restrict__ f1, const float4* __restrict__ f2,
    const float4* __restrict__ f3, const float4* __restrict__ f4,
    uint2* __restrict__ sOut) {
  int i = blockIdx.x * 256 + threadIdx.x;     // < B*512/4
  float4 a = f1[i], b = f2[i], c = f3[i], d = f4[i];
  float x = 0.25f * (a.x + b.x + c.x + d.x);
  float y = 0.25f * (a.y + b.y + c.y + d.y);
  float z = 0.25f * (a.z + b.z + c.z + d.z);
  float w = 0.25f * (a.w + b.w + c.w + d.w);
  uint2 o;
  o.x = (unsigned)f2bf(x) | ((unsigned)f2bf(y) << 16);
  o.y = (unsigned)f2bf(z) | ((unsigned)f2bf(w) << 16);
  sOut[i] = o;
}

// ---------------------------------------------------------------------------
// t = s @ Wg^T(+bg) -> bf16.  Block 64(M) x 256(N), K=512 in 16 steps of 32.
// 8 waves as 2(M) x 4(N); each wave 32x64 = 2x4 WMMA subtiles.
// Register-prefetch double-buffered LDS (proven pipelined codegen).
// ---------------------------------------------------------------------------
__global__ __launch_bounds__(256) void gemm_t_kernel(
    const unsigned short* __restrict__ Asrc,  // [B,512] bf16
    const unsigned short* __restrict__ Wt,    // [512 N][512 K] bf16
    const float* __restrict__ bias,           // [512]
    unsigned short* __restrict__ outB) {      // [B,512] bf16
  __shared__ __align__(16) unsigned short At[2][64][32];
  __shared__ __align__(16) unsigned short Bt[2][256][32];

  const int tid    = threadIdx.x;
  const int lane   = tid & 31;
  const int wave   = tid >> 5;
  const int waveM  = wave >> 2;
  const int waveN  = wave & 3;
  const int lane15 = lane & 15;
  const int khalf  = lane >> 4;

  const int rowBase = blockIdx.x * 64;
  const int colBase = blockIdx.y * 256;

  const int ar = tid >> 2;
  const int ac = (tid & 3) * 8;
  const size_t aRow = (size_t)(rowBase + ar) * 512;
  const size_t bRow = (size_t)(colBase + tid) * 512;

  v8f acc[2][4];
#pragma unroll
  for (int mi = 0; mi < 2; ++mi)
#pragma unroll
    for (int ni = 0; ni < 4; ++ni)
      acc[mi][ni] = (v8f){0.f, 0.f, 0.f, 0.f, 0.f, 0.f, 0.f, 0.f};

  uint4 aR  = *(const uint4*)&Asrc[aRow + ac];
  uint4 bR0 = *(const uint4*)&Wt[bRow + 0];
  uint4 bR1 = *(const uint4*)&Wt[bRow + 8];
  uint4 bR2 = *(const uint4*)&Wt[bRow + 16];
  uint4 bR3 = *(const uint4*)&Wt[bRow + 24];
  *(uint4*)&At[0][ar][ac] = aR;
  *(uint4*)&Bt[0][tid][0]  = bR0;
  *(uint4*)&Bt[0][tid][8]  = bR1;
  *(uint4*)&Bt[0][tid][16] = bR2;
  *(uint4*)&Bt[0][tid][24] = bR3;
  __syncthreads();

  int buf = 0;
#pragma unroll 1
  for (int kt = 0; kt < 16; ++kt) {
    const int k0n = (kt + 1) * 32;
    if (kt < 15) {
      aR  = *(const uint4*)&Asrc[aRow + k0n + ac];
      bR0 = *(const uint4*)&Wt[bRow + k0n + 0];
      bR1 = *(const uint4*)&Wt[bRow + k0n + 8];
      bR2 = *(const uint4*)&Wt[bRow + k0n + 16];
      bR3 = *(const uint4*)&Wt[bRow + k0n + 24];
    }
    FragBF a[2], b[4];
#pragma unroll
    for (int mi = 0; mi < 2; ++mi) {
      const int m = waveM * 32 + mi * 16 + lane15;
      a[mi].u[0] = *(const uint4*)&At[buf][m][khalf * 8];
      a[mi].u[1] = *(const uint4*)&At[buf][m][16 + khalf * 8];
    }
#pragma unroll
    for (int ni = 0; ni < 4; ++ni) {
      const int n = waveN * 64 + ni * 16 + lane15;
      b[ni].u[0] = *(const uint4*)&Bt[buf][n][khalf * 8];
      b[ni].u[1] = *(const uint4*)&Bt[buf][n][16 + khalf * 8];
    }
#pragma unroll
    for (int mi = 0; mi < 2; ++mi)
#pragma unroll
      for (int ni = 0; ni < 4; ++ni)
        acc[mi][ni] = __builtin_amdgcn_wmma_f32_16x16x32_bf16(
            false, a[mi].v, false, b[ni].v, (short)0, acc[mi][ni], false, false);
    if (kt < 15) {
      *(uint4*)&At[buf ^ 1][ar][ac] = aR;
      *(uint4*)&Bt[buf ^ 1][tid][0]  = bR0;
      *(uint4*)&Bt[buf ^ 1][tid][8]  = bR1;
      *(uint4*)&Bt[buf ^ 1][tid][16] = bR2;
      *(uint4*)&Bt[buf ^ 1][tid][24] = bR3;
    }
    __syncthreads();
    buf ^= 1;
  }

#pragma unroll
  for (int mi = 0; mi < 2; ++mi)
#pragma unroll
    for (int ni = 0; ni < 4; ++ni) {
      const int col = colBase + waveN * 64 + ni * 16 + lane15;
      const float bv = bias[col];
#pragma unroll
      for (int r = 0; r < 8; ++r) {
        const int row = rowBase + waveM * 32 + mi * 16 + khalf * 8 + r;
        outB[(size_t)row * 512 + col] = f2bf(acc[mi][ni][r] + bv);
      }
    }
}

// ---------------------------------------------------------------------------
// Fused per-head FC1+FC2: one block = 64 rows x full 512 cols.
//   phase 1: hid = relu(t @ W1s^T + b1)  -> LDS tile [64][512] bf16
//   phase 2: out = relu(sigmoid(hid @ W2^T + b2)) * fea -> f32 global
// 8 waves as 2(M) x 4(N); each wave 32x128 = 2x8 WMMA subtiles (128 acc VGPRs).
// B tiles (512x32 bf16, 32KB) double-buffered, staged async to LDS.
// LDS: 8 + 64 + 64 = 136 KB.
// ---------------------------------------------------------------------------
__global__ __launch_bounds__(256) void hid_gate_kernel(
    const unsigned short* __restrict__ tB,    // [B,512] bf16
    const unsigned short* __restrict__ W1t,   // [512 N][512 K] bf16 (head)
    const unsigned short* __restrict__ W2t,   // [512 N][512 K] bf16 (head)
    const float* __restrict__ b1,             // [512]
    const float* __restrict__ b2,             // [512]
    const float* __restrict__ fea,            // [B,512] f32
    float* __restrict__ outF) {               // [B,512] f32
  __shared__ __align__(16) unsigned short At[2][64][32];    //  8 KiB
  __shared__ __align__(16) unsigned short Bt[2][512][32];   // 64 KiB
  __shared__ __align__(16) unsigned short Hid[64][512];     // 64 KiB

  const int tid    = threadIdx.x;
  const int lane   = tid & 31;
  const int wave   = tid >> 5;
  const int waveM  = wave >> 2;    // 0..1 -> M offset 0/32
  const int waveN  = wave & 3;     // 0..3 -> N offset 0/128/256/384
  const int lane15 = lane & 15;
  const int khalf  = lane >> 4;

  const int rowBase = blockIdx.x * 64;
  const int ar = tid >> 2;              // A tile row
  const int ac = (tid & 3) * 8;         // A tile col chunk
  const size_t aRow = (size_t)(rowBase + ar) * 512;

  // --- staging helpers (1 A chunk + 8 B chunks of 16B per thread/k-step) ---
  auto stageA = [&](int kt, int bufi) {
    CP16(&At[bufi][ar][ac], &tB[aRow + kt * 32 + ac]);
  };
  auto stageB = [&](const unsigned short* __restrict__ W, int kt, int bufi) {
#pragma unroll
    for (int j = 0; j < 2; ++j) {
      const int n = tid + j * 256;
      const unsigned short* src = &W[(size_t)n * 512 + kt * 32];
#pragma unroll
      for (int q = 0; q < 4; ++q) CP16(&Bt[bufi][n][q * 8], src + q * 8);
    }
  };

  v8f acc[2][8];
#pragma unroll
  for (int mi = 0; mi < 2; ++mi)
#pragma unroll
    for (int ni = 0; ni < 8; ++ni)
      acc[mi][ni] = (v8f){0.f, 0.f, 0.f, 0.f, 0.f, 0.f, 0.f, 0.f};

  // ============================ phase 1: hid ===============================
  stageA(0, 0);
  stageB(W1t, 0, 0);
  WAIT_STAGE();
  __syncthreads();

  int buf = 0;
#pragma unroll 1
  for (int kt = 0; kt < 16; ++kt) {
    if (kt < 15) { stageA(kt + 1, buf ^ 1); stageB(W1t, kt + 1, buf ^ 1); }
    FragBF a[2];
#pragma unroll
    for (int mi = 0; mi < 2; ++mi) {
      const int m = waveM * 32 + mi * 16 + lane15;
      a[mi].u[0] = *(const uint4*)&At[buf][m][khalf * 8];
      a[mi].u[1] = *(const uint4*)&At[buf][m][16 + khalf * 8];
    }
#pragma unroll
    for (int ni = 0; ni < 8; ++ni) {
      const int n = waveN * 128 + ni * 16 + lane15;
      FragBF bf;
      bf.u[0] = *(const uint4*)&Bt[buf][n][khalf * 8];
      bf.u[1] = *(const uint4*)&Bt[buf][n][16 + khalf * 8];
#pragma unroll
      for (int mi = 0; mi < 2; ++mi)
        acc[mi][ni] = __builtin_amdgcn_wmma_f32_16x16x32_bf16(
            false, a[mi].v, false, bf.v, (short)0, acc[mi][ni], false, false);
    }
    WAIT_STAGE();
    __syncthreads();
    buf ^= 1;
  }

  // prefetch first W2 tile while writing hid to LDS (Bt[0] is free now)
  stageB(W2t, 0, 0);

  // hid = relu(acc + b1) -> LDS bf16, then reset accumulators
#pragma unroll
  for (int mi = 0; mi < 2; ++mi)
#pragma unroll
    for (int ni = 0; ni < 8; ++ni) {
      const int n = waveN * 128 + ni * 16 + lane15;
      const float bv = b1[n];
#pragma unroll
      for (int r = 0; r < 8; ++r) {
        const int m = waveM * 32 + mi * 16 + khalf * 8 + r;
        Hid[m][n] = f2bf(fmaxf(acc[mi][ni][r] + bv, 0.f));
      }
      acc[mi][ni] = (v8f){0.f, 0.f, 0.f, 0.f, 0.f, 0.f, 0.f, 0.f};
    }
  WAIT_STAGE();
  __syncthreads();

  // ============================ phase 2: gate ==============================
  buf = 0;
#pragma unroll 1
  for (int kt = 0; kt < 16; ++kt) {
    if (kt < 15) stageB(W2t, kt + 1, buf ^ 1);
    const int k0 = kt * 32;
    FragBF a[2];
#pragma unroll
    for (int mi = 0; mi < 2; ++mi) {
      const int m = waveM * 32 + mi * 16 + lane15;
      a[mi].u[0] = *(const uint4*)&Hid[m][k0 + khalf * 8];
      a[mi].u[1] = *(const uint4*)&Hid[m][k0 + 16 + khalf * 8];
    }
#pragma unroll
    for (int ni = 0; ni < 8; ++ni) {
      const int n = waveN * 128 + ni * 16 + lane15;
      FragBF bf;
      bf.u[0] = *(const uint4*)&Bt[buf][n][khalf * 8];
      bf.u[1] = *(const uint4*)&Bt[buf][n][16 + khalf * 8];
#pragma unroll
      for (int mi = 0; mi < 2; ++mi)
        acc[mi][ni] = __builtin_amdgcn_wmma_f32_16x16x32_bf16(
            false, a[mi].v, false, bf.v, (short)0, acc[mi][ni], false, false);
    }
    WAIT_STAGE();
    __syncthreads();
    buf ^= 1;
  }

  // out = relu(sigmoid(acc + b2)) * fea
#pragma unroll
  for (int mi = 0; mi < 2; ++mi)
#pragma unroll
    for (int ni = 0; ni < 8; ++ni) {
      const int n = waveN * 128 + ni * 16 + lane15;
      const float bv = b2[n];
#pragma unroll
      for (int r = 0; r < 8; ++r) {
        const int row = rowBase + waveM * 32 + mi * 16 + khalf * 8 + r;
        float g = 1.f / (1.f + __expf(-(acc[mi][ni][r] + bv)));
        g = fmaxf(g, 0.f);  // relu-on-sigmoid kept for faithfulness
        outF[(size_t)row * 512 + n] = g * fea[(size_t)row * 512 + n];
      }
    }
}

// ---------------------------------------------------------------------------
extern "C" void kernel_launch(void* const* d_in, const int* in_sizes, int n_in,
                              void* d_out, int out_size, void* d_ws,
                              size_t ws_size, hipStream_t stream) {
  (void)in_sizes; (void)n_in; (void)out_size; (void)ws_size;

  const float* fea[4] = {(const float*)d_in[0], (const float*)d_in[1],
                         (const float*)d_in[2], (const float*)d_in[3]};
  const float* Wg = (const float*)d_in[4];
  const float* bg = (const float*)d_in[5];
  const float* W1 = (const float*)d_in[6];
  const float* b1 = (const float*)d_in[7];
  const float* W2 = (const float*)d_in[8];
  const float* b2 = (const float*)d_in[9];

  // workspace layout (~38 MiB)
  char* ws = (char*)d_ws;
  unsigned short* sB   = (unsigned short*)(ws);                             // 16 MiB
  unsigned short* tB   = (unsigned short*)(ws + (size_t)16 * 1024 * 1024);  // 16 MiB
  unsigned short* Wgb  = (unsigned short*)(ws + (size_t)32 * 1024 * 1024);  // 0.5 MiB
  unsigned short* W1sb = (unsigned short*)(ws + (size_t)33 * 1024 * 1024);  // 2 MiB
  unsigned short* W2b  = (unsigned short*)(ws + (size_t)36 * 1024 * 1024);  // 2 MiB

  // 1) fold + transpose + cast weights
  prep_weights_kernel<<<9216, 256, 0, stream>>>(Wg, W1, W2, Wgb, W1sb, W2b);

  // 2) s = bf16(0.25 * sum(fea_i))
  mix_kernel<<<(Bsz * C / 4) / 256, 256, 0, stream>>>(
      (const float4*)fea[0], (const float4*)fea[1], (const float4*)fea[2],
      (const float4*)fea[3], (uint2*)sB);

  // 3) t = s @ Wg + bg
  dim3 gridT(Bsz / 64, C / 256, 1);
  gemm_t_kernel<<<gridT, 256, 0, stream>>>(sB, Wgb, bg, tB);

  // 4+5) per head: fused hid -> gated output (hid stays in LDS)
  for (int k = 0; k < 4; ++k)
    hid_gate_kernel<<<Bsz / 64, 256, 0, stream>>>(
        tB, W1sb + (size_t)k * 262144, W2b + (size_t)k * 262144,
        b1 + k * 512, b2 + k * 512, fea[k],
        (float*)d_out + (size_t)k * Bsz * C);
}
